// TK_16260746183005
// MI455X (gfx1250) — matrix operand order repaired and checked by reading, not compiled, not run on hardware
//
#include <hip/hip_runtime.h>
#include <math.h>

typedef _Float16 h16v __attribute__((ext_vector_type(16)));
typedef _Float16 h8v  __attribute__((ext_vector_type(8)));
typedef float    f8v  __attribute__((ext_vector_type(8)));
typedef float    f2v  __attribute__((ext_vector_type(2)));
typedef int      v4i  __attribute__((ext_vector_type(4)));

#define DM   256
#define NH   8
#define DH   32
#define DFF  2048
#define NKER 11
#define NB   256
#define SQ   32
#define SD   256

#if defined(__has_builtin)
#if __has_builtin(__builtin_amdgcn_wmma_f32_16x16x4_f32)
#define HAVE_WMMA_F32 1
#endif
#if __has_builtin(__builtin_amdgcn_global_load_async_to_lds_b128)
#define HAVE_ASYNC_LDS_BUILTIN 1
#endif
#if __has_builtin(__builtin_amdgcn_s_wait_asynccnt)
#define HAVE_WAIT_ASYNC_BUILTIN 1
#endif
#endif

__device__ __forceinline__ void async_wait0() {
#ifdef HAVE_WAIT_ASYNC_BUILTIN
  __builtin_amdgcn_s_wait_asynccnt(0);
#else
  asm volatile("s_wait_asynccnt 0x0" ::: "memory");
#endif
}

// async global -> LDS copy of one 16B chunk (per-thread), tracked by ASYNCcnt
__device__ __forceinline__ void async_copy16(const _Float16* g, _Float16* l) {
#ifdef HAVE_ASYNC_LDS_BUILTIN
  __builtin_amdgcn_global_load_async_to_lds_b128(
      (__attribute__((address_space(1))) v4i*)g,
      (__attribute__((address_space(3))) v4i*)l, 0, 0);
#else
  unsigned la = (unsigned)(unsigned long long)(__attribute__((address_space(3))) _Float16*)l;
  unsigned long long ga = (unsigned long long)g;
  asm volatile("global_load_async_to_lds_b128 %0, %1, off" :: "v"(la), "v"(ga) : "memory");
#endif
}

// ---------------- WMMA fragment loaders (CDNA5 16x16x32 f16 layout) ----------------
// A (16x32 MxK): lane<16 -> row M=lane, K=k0+{0..7},{16..23}; lane>=16 -> K=k0+{8..15},{24..31}
__device__ __forceinline__ h16v load_a16(const _Float16* X, int ldk, int m0, int k0, int lane) {
  int m    = m0 + (lane & 15);
  int koff = k0 + ((lane & 16) ? 8 : 0);
  const _Float16* p = X + (size_t)m * ldk + koff;
  h8v lo = *(const h8v*)p;
  h8v hi = *(const h8v*)(p + 16);
  h16v a;
#pragma unroll
  for (int i = 0; i < 8; ++i) { a[i] = lo[i]; a[i + 8] = hi[i]; }
  return a;
}

// A fragment directly from a lane-relative pointer (pointer-increment addressing)
__device__ __forceinline__ h16v load_a16_ptr(const _Float16* p) {
  h8v lo = *(const h8v*)p;
  h8v hi = *(const h8v*)(p + 16);
  h16v a;
#pragma unroll
  for (int i = 0; i < 8; ++i) { a[i] = lo[i]; a[i + 8] = hi[i]; }
  return a;
}

// B (32x16 KxN): lane<16 -> col N=n0+lane, K=k0+{0..7},{16..23}; lane>=16 -> K=k0+{8..15},{24..31}
__device__ __forceinline__ h16v load_b16(const _Float16* Wt, int ldn, int k0, int n0, int lane) {
  const _Float16* p = Wt + (size_t)(k0 + ((lane & 16) ? 8 : 0)) * ldn + n0 + (lane & 15);
  h16v b;
#pragma unroll
  for (int i = 0; i < 8; ++i) { b[i] = p[(size_t)i * ldn]; b[i + 8] = p[(size_t)(16 + i) * ldn]; }
  return b;
}

// B fragment from an LDS panel with padded row stride (column gather, table K order)
__device__ __forceinline__ h16v load_b_lds(const _Float16* buf, int stride, int n0, int lane) {
  const _Float16* p = buf + ((lane & 16) ? 8 : 0) * stride + n0 + (lane & 15);
  h16v b;
#pragma unroll
  for (int i = 0; i < 8; ++i) { b[i] = p[i * stride]; b[i + 8] = p[(16 + i) * stride]; }
  return b;
}

__device__ __forceinline__ float wave_sum(float v) {
#pragma unroll
  for (int o = 16; o > 0; o >>= 1) v += __shfl_xor(v, o, 32);
  return v;
}
__device__ __forceinline__ float wave_max(float v) {
#pragma unroll
  for (int o = 16; o > 0; o >>= 1) v = fmaxf(v, __shfl_xor(v, o, 32));
  return v;
}

// ---------------- small utility kernels ----------------
__global__ void zero_out(float* out, int n) {
  int i = blockIdx.x * blockDim.x + threadIdx.x;
  if (i < n) out[i] = 0.f;
}

// W[N,K] f32 -> Wt[K,N] f16
__global__ void wcvt(const float* __restrict__ W, _Float16* __restrict__ Wt, int N, int K) {
  size_t i = (size_t)blockIdx.x * blockDim.x + threadIdx.x;
  if (i >= (size_t)N * K) return;
  int n = (int)(i / K);
  int k = (int)(i % K);
  Wt[(size_t)k * N + n] = (_Float16)W[i];
}

// gather embedding, apply mask, add PE: x = 2*emb[id] + pe ; save masked emb (f32)
__global__ void embed_pe(const int* __restrict__ ids, const float* __restrict__ emb,
                         float* __restrict__ emb32, float* __restrict__ x32,
                         _Float16* __restrict__ x16, int B, int S) {
  size_t i = (size_t)blockIdx.x * blockDim.x + threadIdx.x;
  size_t total = (size_t)B * S * DM;
  if (i >= total) return;
  int e = (int)(i & (DM - 1));
  size_t row = i >> 8;
  int sidx = (int)(row % S);
  int id = ids[row];
  float mask = (id > 0) ? 1.f : 0.f;
  float v = emb[(size_t)id * DM + e] * mask;
  emb32[i] = v;
  int i2 = e & ~1;
  float div = __expf(-logf(10000.f) * (float)i2 / (float)DM);
  float ang = (float)sidx * div;
  float pe = (e & 1) ? cosf(ang) : sinf(ang);
  float x = 2.f * v + pe;
  x32[i] = x;
  x16[i] = (_Float16)x;
}

// ---------------- WMMA GEMM with async-LDS double-buffered B panel ----------------
// Y = act(X @ Wt + bias). X f16 [M,K], Wt f16 [K,N] (pre-transposed weights).
// Block = 8 waves; block tile 128M x 64N; per k-step the 32x64 B panel is staged
// into LDS once via GLOBAL_LOAD_ASYNC_TO_LDS_B128 and shared by all 8 waves.
#define BPAD 72  // padded LDS row stride in f16 (144B, 16B aligned, spreads banks)

__global__ __launch_bounds__(256) void gemm_wmma(
    const _Float16* __restrict__ X, const _Float16* __restrict__ Wt,
    const float* __restrict__ bias, _Float16* __restrict__ Y16,
    float* __restrict__ Y32, int M, int N, int K, int relu) {
  __shared__ _Float16 bpan[2][32 * BPAD];
  int tid  = threadIdx.x;
  int lane = tid & 31;
  int wv   = tid >> 5;
  int tn64 = N >> 6;
  int bid  = blockIdx.x;
  int m0 = (bid / tn64) * 128 + wv * 16;
  int n0 = (bid % tn64) * 64;

  // B staging: 256 threads x 16B = 4KB = one 32x64 f16 panel
  const int srow = tid >> 3;         // 0..31 (k within panel)
  const int scol = (tid & 7) << 3;   // f16 column within panel
  const _Float16* gB = Wt + (size_t)srow * N + n0 + scol;  // advances 32*N per step

  // A operand: lane-relative streaming pointer
  const _Float16* pa = X + (size_t)(m0 + (lane & 15)) * K + ((lane & 16) ? 8 : 0);

  // prologue: stage panel 0
  async_copy16(gB, &bpan[0][srow * BPAD + scol]);

  f8v acc[4] = {f8v{}, f8v{}, f8v{}, f8v{}};
  int nsteps = K >> 5;
  for (int s = 0; s < nsteps; ++s) {
    async_wait0();
    __syncthreads();                 // panel s visible to all waves
    if (s + 1 < nsteps)              // stage next panel while computing
      async_copy16(gB + (size_t)(s + 1) * 32 * N, &bpan[(s + 1) & 1][srow * BPAD + scol]);
    h16v a = load_a16_ptr(pa);
    pa += 32;
    const _Float16* bp = bpan[s & 1];
#pragma unroll
    for (int t = 0; t < 4; ++t) {
      h16v b = load_b_lds(bp, BPAD, t * 16, lane);
      acc[t] = __builtin_amdgcn_wmma_f32_16x16x32_f16(false, a, false, b,
                                                      (short)0, acc[t], false, false);
    }
    __syncthreads();                 // all waves done reading panel s
  }

  int nlo = lane & 15;
  int mhi = (lane & 16) ? 8 : 0;
#pragma unroll
  for (int t = 0; t < 4; ++t) {
#pragma unroll
    for (int r = 0; r < 8; ++r) {
      int n = n0 + t * 16 + nlo;
      int m = m0 + mhi + r;
      float v = acc[t][r] + bias[n];
      if (relu) v = fmaxf(v, 0.f);
      if (Y16) Y16[(size_t)m * N + n] = (_Float16)v;
      if (Y32) Y32[(size_t)m * N + n] = v;
    }
  }
}

// split packed qkv [M,768] into Q[B,H,S,32], K^T[B,H,32,S], V[B,H,S,32]
__global__ void split_heads(const _Float16* __restrict__ qkv, _Float16* __restrict__ Qh,
                            _Float16* __restrict__ KT, _Float16* __restrict__ Vh,
                            int B, int S) {
  size_t i = (size_t)blockIdx.x * blockDim.x + threadIdx.x;
  size_t total = (size_t)B * S * DM;
  if (i >= total) return;
  int e = (int)(i & (DM - 1));
  size_t row = i >> 8;
  int sidx = (int)(row % S);
  int b = (int)(row / S);
  int h = e >> 5;
  int d = e & 31;
  const _Float16* src = qkv + row * (3 * DM);
  size_t base = (size_t)(b * NH + h);
  Qh[base * S * DH + (size_t)sidx * DH + d] = src[e];
  KT[base * DH * S + (size_t)d * S + sidx] = src[DM + e];
  Vh[base * S * DH + (size_t)sidx * DH + d] = src[2 * DM + e];
}

// fused attention: one wave handles (b, h, 16-query-row tile); scores+softmax in LDS
__global__ void attn_wmma(const _Float16* __restrict__ Qh, const _Float16* __restrict__ KT,
                          const _Float16* __restrict__ Vh, _Float16* __restrict__ O16,
                          int B, int S) {
  extern __shared__ float sm[];
  int lane = threadIdx.x & 31;
  int wib  = threadIdx.x >> 5;
  float* sc = sm + (size_t)wib * 16 * S;
  int wid = blockIdx.x * (blockDim.x >> 5) + wib;
  int qtiles = S >> 4;
  if (wid >= B * NH * qtiles) return;
  int qt = wid % qtiles;
  int bh = wid / qtiles;
  int h = bh % NH, b = bh / NH;
  const _Float16* Q  = Qh + (size_t)(b * NH + h) * S * DH;
  const _Float16* Kt = KT + (size_t)(b * NH + h) * DH * S;
  const _Float16* V  = Vh + (size_t)(b * NH + h) * S * DH;
  int m0 = qt << 4;
  int nlo = lane & 15, mhi = (lane & 16) ? 8 : 0;
  const float scale = 0.1767766952966369f; // 1/sqrt(32)

  // scores = Q K^T / sqrt(dh)
  h16v a = load_a16(Q, DH, m0, 0, lane);
  for (int nt = 0; nt < (S >> 4); ++nt) {
    h16v bf = load_b16(Kt, S, 0, nt * 16, lane);
    f8v c = {};
    c = __builtin_amdgcn_wmma_f32_16x16x32_f16(false, a, false, bf, (short)0, c, false, false);
#pragma unroll
    for (int r = 0; r < 8; ++r) sc[(size_t)(mhi + r) * S + nt * 16 + nlo] = c[r] * scale;
  }

  // softmax rows (wave-cooperative)
  for (int r = 0; r < 16; ++r) {
    float* row = sc + (size_t)r * S;
    float mx = -1e30f;
    for (int cc = lane; cc < S; cc += 32) mx = fmaxf(mx, row[cc]);
    mx = wave_max(mx);
    float sum = 0.f;
    for (int cc = lane; cc < S; cc += 32) { float e = __expf(row[cc] - mx); row[cc] = e; sum += e; }
    sum = wave_sum(sum);
    float inv = 1.f / sum;
    for (int cc = lane; cc < S; cc += 32) row[cc] *= inv;
  }

  // out = attn @ V
  f8v o0 = {}, o1 = {};
  for (int k0 = 0; k0 < S; k0 += 32) {
    int am = lane & 15;
    int koff = k0 + ((lane & 16) ? 8 : 0);
    h16v aa;
#pragma unroll
    for (int i = 0; i < 8; ++i) {
      aa[i]     = (_Float16)sc[(size_t)am * S + koff + i];
      aa[i + 8] = (_Float16)sc[(size_t)am * S + koff + 16 + i];
    }
    h16v b0 = load_b16(V, DH, k0, 0, lane);
    h16v b1 = load_b16(V, DH, k0, 16, lane);
    o0 = __builtin_amdgcn_wmma_f32_16x16x32_f16(false, aa, false, b0, (short)0, o0, false, false);
    o1 = __builtin_amdgcn_wmma_f32_16x16x32_f16(false, aa, false, b1, (short)0, o1, false, false);
  }
  _Float16* Ob = O16 + (size_t)b * S * DM + h * DH;
#pragma unroll
  for (int r = 0; r < 8; ++r) {
    Ob[(size_t)(m0 + mhi + r) * DM + nlo]      = (_Float16)o0[r];
    Ob[(size_t)(m0 + mhi + r) * DM + 16 + nlo] = (_Float16)o1[r];
  }
}

// x' = LN(x + y) * s + b ; writes f32 (residual chain) and f16 (next GEMM A operand)
__global__ void ln_residual(const float* __restrict__ X, const float* __restrict__ Yd,
                            const float* __restrict__ s, const float* __restrict__ bsh,
                            float* __restrict__ Xo, _Float16* __restrict__ Xo16, int M) {
  int lane = threadIdx.x & 31;
  int row = (int)(((size_t)blockIdx.x * blockDim.x + threadIdx.x) >> 5);
  if (row >= M) return;
  const float* x = X + (size_t)row * DM;
  const float* y = Yd + (size_t)row * DM;
  float v[8], s1 = 0.f, s2 = 0.f;
#pragma unroll
  for (int i = 0; i < 8; ++i) {
    v[i] = x[lane + i * 32] + y[lane + i * 32];
    s1 += v[i]; s2 += v[i] * v[i];
  }
  s1 = wave_sum(s1); s2 = wave_sum(s2);
  float mean = s1 * (1.f / DM);
  float var = s2 * (1.f / DM) - mean * mean;
  float inv = rsqrtf(var + 1e-5f);
#pragma unroll
  for (int i = 0; i < 8; ++i) {
    int c = lane + i * 32;
    float o = (v[i] - mean) * inv * s[c] + bsh[c];
    Xo[(size_t)row * DM + c] = o;
    Xo16[(size_t)row * DM + c] = (_Float16)o;
  }
}

// mix = (a*emb + (1-a)*ctx) * mask
__global__ void mix_k(const int* __restrict__ ids, const float* __restrict__ emb32,
                      const float* __restrict__ ctx32, const float* __restrict__ aP,
                      float* __restrict__ mix, int B, int S) {
  size_t i = (size_t)blockIdx.x * blockDim.x + threadIdx.x;
  if (i >= (size_t)B * S * DM) return;
  size_t row = i >> 8;
  float mask = (ids[row] > 0) ? 1.f : 0.f;
  float a = aP[0];
  mix[i] = (a * emb32[i] + (1.f - a) * ctx32[i]) * mask;
}

// sim GEMM (f32 WMMA) + RBF kernel pooling + score reduction. One wave per (b, qtile).
__global__ void sim_knrm(const float* __restrict__ qmix, const float* __restrict__ dmix,
                         const int* __restrict__ qids, const int* __restrict__ dids,
                         const float* __restrict__ mlpw, float* __restrict__ out) {
  extern __shared__ float sm[];
  int lane = threadIdx.x & 31;
  int wib = threadIdx.x >> 5;
  float* sc = sm + (size_t)wib * 16 * SD;
  int wid = blockIdx.x * (blockDim.x >> 5) + wib;
  if (wid >= NB * 2) return;
  int qt = wid & 1, b = wid >> 1;
  const float* Qm = qmix + (size_t)b * SQ * DM + (size_t)(qt * 16) * DM;
  const float* Dm = dmix + (size_t)b * SD * DM;
  int nlo = lane & 15, mhi = (lane & 16) ? 8 : 0;

#ifdef HAVE_WMMA_F32
  int koffA = (lane & 16) ? 2 : 0;
  for (int nt = 0; nt < 16; ++nt) {
    f8v c = {};
    for (int k0 = 0; k0 < DM; k0 += 4) {
      f2v a, bb;
      a.x = Qm[(size_t)(lane & 15) * DM + k0 + koffA];
      a.y = Qm[(size_t)(lane & 15) * DM + k0 + koffA + 1];
      bb.x = Dm[(size_t)(nt * 16 + nlo) * DM + k0 + koffA];
      bb.y = Dm[(size_t)(nt * 16 + nlo) * DM + k0 + koffA + 1];
      c = __builtin_amdgcn_wmma_f32_16x16x4_f32(false, a, false, bb, (short)0, c, false, false);
    }
#pragma unroll
    for (int r = 0; r < 8; ++r) sc[(size_t)(mhi + r) * SD + nt * 16 + nlo] = c[r];
  }
#else
  for (int idx = lane; idx < 16 * SD; idx += 32) {
    int m = idx >> 8, cc = idx & 255;
    float acc = 0.f;
    for (int k = 0; k < DM; ++k) acc += Qm[(size_t)m * DM + k] * Dm[(size_t)cc * DM + k];
    sc[(size_t)m * SD + cc] = acc;
  }
#endif

  float mu[NKER], i2s[NKER];
  mu[0] = 1.f;
  float bin = 2.f / (NKER - 1);
  mu[1] = 1.f - bin * 0.5f;
  for (int i = 2; i < NKER; ++i) mu[i] = mu[i - 1] - bin;
  i2s[0] = 1.f / (2.f * 1e-4f * 1e-4f);
  for (int i = 1; i < NKER; ++i) { float sg = 0.5f * bin; i2s[i] = 1.f / (2.f * sg * sg); }

  float total = 0.f;
  for (int r = 0; r < 16; ++r) {
    int q = qt * 16 + r;
    float qmask = (qids[b * SQ + q] > 0) ? 1.f : 0.f;
    float part[NKER];
#pragma unroll
    for (int k = 0; k < NKER; ++k) part[k] = 0.f;
    for (int cc = lane; cc < SD; cc += 32) {
      float dmask = (dids[b * SD + cc] > 0) ? 1.f : 0.f;
      float sv = sc[(size_t)r * SD + cc];
#pragma unroll
      for (int k = 0; k < NKER; ++k) {
        float d = sv - mu[k];
        part[k] += dmask * __expf(-d * d * i2s[k]);
      }
    }
#pragma unroll
    for (int k = 0; k < NKER; ++k) part[k] = wave_sum(part[k]);
    if (lane == 0) {
#pragma unroll
      for (int k = 0; k < NKER; ++k) {
        float qk = part[k] * qmask;
        total += mlpw[k] * (qmask * __logf(fmaxf(qk, 1e-10f)) + qk * (1.f / SD));
      }
    }
  }
  if (lane == 0) atomicAdd(&out[b], total);
}

// ---------------- host orchestration ----------------
extern "C" void kernel_launch(void* const* d_in, const int* in_sizes, int n_in,
                              void* d_out, int out_size, void* d_ws, size_t ws_size,
                              hipStream_t stream) {
  (void)in_sizes; (void)n_in; (void)out_size; (void)ws_size;
  const int*   q_ids = (const int*)  d_in[0];
  const int*   d_idsp= (const int*)  d_in[1];
  const float* emb   = (const float*)d_in[2];
  const float* aP    = (const float*)d_in[3];
  const float* mlpw  = (const float*)d_in[4];
  const float* Wqkv  = (const float*)d_in[5];
  const float* bqkv  = (const float*)d_in[6];
  const float* Wo    = (const float*)d_in[7];
  const float* bo    = (const float*)d_in[8];
  const float* ln1s  = (const float*)d_in[9];
  const float* ln1b  = (const float*)d_in[10];
  const float* W1    = (const float*)d_in[11];
  const float* b1    = (const float*)d_in[12];
  const float* W2    = (const float*)d_in[13];
  const float* b2    = (const float*)d_in[14];
  const float* ln2s  = (const float*)d_in[15];
  const float* ln2b  = (const float*)d_in[16];

  char* p = (char*)d_ws;
  auto alloc = [&](size_t bytes) -> char* {
    char* r = p; p += (bytes + 255) & ~(size_t)255; return r;
  };
  const size_t MQ = (size_t)NB * SQ;   // 8192
  const size_t MD = (size_t)NB * SD;   // 65536

  _Float16 *wt_qkv[2], *wt_o[2], *wt_1[2], *wt_2[2];
  for (int l = 0; l < 2; ++l) {
    wt_qkv[l] = (_Float16*)alloc(sizeof(_Float16) * 256 * 768);
    wt_o[l]   = (_Float16*)alloc(sizeof(_Float16) * 256 * 256);
    wt_1[l]   = (_Float16*)alloc(sizeof(_Float16) * 256 * 2048);
    wt_2[l]   = (_Float16*)alloc(sizeof(_Float16) * 2048 * 256);
  }
  float*     q_emb32 = (float*)alloc(MQ * DM * 4);
  float*     d_emb32 = (float*)alloc(MD * DM * 4);
  float*     q_x32   = (float*)alloc(MQ * DM * 4);
  float*     d_x32   = (float*)alloc(MD * DM * 4);
  _Float16*  q_x16   = (_Float16*)alloc(MQ * DM * 2);
  _Float16*  d_x16   = (_Float16*)alloc(MD * DM * 2);
  _Float16*  t_qkv   = (_Float16*)alloc(MD * 768 * 2);
  _Float16*  t_Qh    = (_Float16*)alloc(MD * DM * 2);
  _Float16*  t_KT    = (_Float16*)alloc(MD * DM * 2);
  _Float16*  t_Vh    = (_Float16*)alloc(MD * DM * 2);
  _Float16*  t_o16   = (_Float16*)alloc(MD * DM * 2);
  float*     t_y32   = (float*)alloc(MD * DM * 4);
  _Float16*  t_h16   = (_Float16*)alloc(MD * DFF * 2);
  float*     q_mix   = (float*)alloc(MQ * DM * 4);
  float*     d_mix   = (float*)alloc(MD * DM * 4);

  zero_out<<<1, 256, 0, stream>>>((float*)d_out, NB);

  auto cvt = [&](const float* W, _Float16* Wt, int N, int K) {
    size_t tot = (size_t)N * K;
    wcvt<<<(int)((tot + 255) / 256), 256, 0, stream>>>(W, Wt, N, K);
  };
  for (int l = 0; l < 2; ++l) {
    cvt(Wqkv + (size_t)l * 768 * 256, wt_qkv[l], 768, 256);
    cvt(Wo   + (size_t)l * 256 * 256, wt_o[l],   256, 256);
    cvt(W1   + (size_t)l * 2048 * 256, wt_1[l],  2048, 256);
    cvt(W2   + (size_t)l * 256 * 2048, wt_2[l],  256, 2048);
  }

  embed_pe<<<(int)((MQ * DM + 255) / 256), 256, 0, stream>>>(q_ids, emb, q_emb32, q_x32, q_x16, NB, SQ);
  embed_pe<<<(int)((MD * DM + 255) / 256), 256, 0, stream>>>(d_idsp, emb, d_emb32, d_x32, d_x16, NB, SD);

  auto gemm = [&](const _Float16* X, const _Float16* Wt, const float* bias,
                  _Float16* Y16, float* Y32, int M, int N, int K, int relu) {
    int blocks = (M / 128) * (N / 64);   // all dims divide exactly
    gemm_wmma<<<blocks, 256, 0, stream>>>(X, Wt, bias, Y16, Y32, M, N, K, relu);
  };

  auto run_encoder = [&](float* x32, _Float16* x16, int S) {
    int M = NB * S;
    for (int l = 0; l < 2; ++l) {
      gemm(x16, wt_qkv[l], bqkv + (size_t)l * 768, t_qkv, nullptr, M, 768, 256, 0);
      split_heads<<<(int)(((size_t)M * DM + 255) / 256), 256, 0, stream>>>(t_qkv, t_Qh, t_KT, t_Vh, NB, S);
      {
        int waves = NB * NH * (S / 16);
        const int wpb = 4;
        int blocks = (waves + wpb - 1) / wpb;
        size_t shmem = (size_t)wpb * 16 * S * sizeof(float);
        attn_wmma<<<blocks, wpb * 32, shmem, stream>>>(t_Qh, t_KT, t_Vh, t_o16, NB, S);
      }
      gemm(t_o16, wt_o[l], bo + (size_t)l * 256, nullptr, t_y32, M, 256, 256, 0);
      ln_residual<<<(M + 7) / 8, 256, 0, stream>>>(x32, t_y32, ln1s + (size_t)l * 256, ln1b + (size_t)l * 256, x32, x16, M);
      gemm(x16, wt_1[l], b1 + (size_t)l * 2048, t_h16, nullptr, M, 2048, 256, 1);
      gemm(t_h16, wt_2[l], b2 + (size_t)l * 256, nullptr, t_y32, M, 256, 2048, 0);
      ln_residual<<<(M + 7) / 8, 256, 0, stream>>>(x32, t_y32, ln2s + (size_t)l * 256, ln2b + (size_t)l * 256, x32, x16, M);
    }
  };
  run_encoder(q_x32, q_x16, SQ);
  run_encoder(d_x32, d_x16, SD);

  mix_k<<<(int)((MQ * DM + 255) / 256), 256, 0, stream>>>(q_ids, q_emb32, q_x32, aP, q_mix, NB, SQ);
  mix_k<<<(int)((MD * DM + 255) / 256), 256, 0, stream>>>(d_idsp, d_emb32, d_x32, aP, d_mix, NB, SD);

  {
    int waves = NB * 2;
    const int wpb = 4;
    size_t shmem = (size_t)wpb * 16 * SD * sizeof(float);
    sim_knrm<<<(waves + wpb - 1) / wpb, wpb * 32, shmem, stream>>>(q_mix, d_mix, q_ids, d_idsp, mlpw, (float*)d_out);
  }
}
